// GNN_75436805587134
// MI455X (gfx1250) — compile-verified, matrix operand order checked by reading, not run on hardware
//
#include <hip/hip_runtime.h>

// ---------------------------------------------------------------------------
// GNN diffusion forward for MI455X (gfx1250, wave32, WMMA).
// Heavy math (6x N=100000 x 256x256 GEMMs) on v_wmma_f32_16x16x32_bf16.
// Edge aggregation via f32 global atomics (memory-bound path at 23.3 TB/s).
// ---------------------------------------------------------------------------

typedef __attribute__((ext_vector_type(16))) __bf16 v16bf;
typedef __attribute__((ext_vector_type(8)))  float  v8f;

#define N_NODES 100000
#define N_EDGES 400000
#define N_BATCH 1024
#define H_      256
#define D_      6
#define LDA     264   // padded LDS row stride (bf16 elems): 264*2/4 = 132 dwords %64 = 4 -> conflict-free

// ---------------- degree / norm kernels ----------------
__global__ void k_deg(const int* __restrict__ src, const int* __restrict__ dst,
                      float* __restrict__ dego, float* __restrict__ degi, int ne) {
    int e = blockIdx.x * 256 + threadIdx.x;
    if (e < ne) {
        atomicAdd(&dego[src[e]], 1.0f);
        atomicAdd(&degi[dst[e]], 1.0f);
    }
}

__global__ void k_norms(const float* __restrict__ dego, const float* __restrict__ degi,
                        float* __restrict__ no_v, float* __restrict__ ni_v, int n) {
    int i = blockIdx.x * 256 + threadIdx.x;
    if (i < n) {
        no_v[i] = rsqrtf(fmaxf(dego[i], 1.0f));
        ni_v[i] = rsqrtf(fmaxf(degi[i], 1.0f));
    }
}

// ---------------- input-stage scatters (linearity: aggregate BEFORE projection) ---
__global__ void k_scat32(const float* __restrict__ x, const float* __restrict__ no_v,
                         const int* __restrict__ src, const int* __restrict__ dst,
                         float* __restrict__ out, int total) {
    int g = blockIdx.x * 256 + threadIdx.x;
    if (g >= total) return;
    int e = g >> 5, f = g & 31;
    int s = src[e];
    atomicAdd(&out[(size_t)dst[e] * 32 + f], x[(size_t)s * 32 + f] * no_v[s]);
}

__global__ void k_scat16(const float* __restrict__ ef, const int* __restrict__ dst,
                         float* __restrict__ out, int total) {
    int g = blockIdx.x * 256 + threadIdx.x;
    if (g >= total) return;
    int e = g >> 4, f = g & 15;
    atomicAdd(&out[(size_t)dst[e] * 16 + f], ef[(size_t)e * 16 + f]);
}

// ---------------- time embedding -> cond (B x 256) ----------------
__global__ void k_cond(const float* __restrict__ t, const float* __restrict__ w1,
                       const float* __restrict__ b1, const float* __restrict__ w2,
                       const float* __restrict__ b2, float* __restrict__ cond) {
    __shared__ float emb[128];
    __shared__ float hid[256];
    int b = blockIdx.x, c = threadIdx.x;
    float tv = t[b] * 1000.0f;
    if (c < 128) {
        int j = c & 63;
        float f = __expf(-6.907755278982137f * (float)j * (1.0f / 64.0f)); // ln(1000)
        float a = tv * f;
        emb[c] = (c < 64) ? sinf(a) : cosf(a);
    }
    __syncthreads();
    float acc = b1[c];
    for (int k = 0; k < 128; ++k) acc += emb[k] * w1[k * H_ + c];
    hid[c] = acc / (1.0f + __expf(-acc));   // SiLU
    __syncthreads();
    float acc2 = b2[c];
    for (int k = 0; k < 256; ++k) acc2 += hid[k] * w2[k * H_ + c];
    cond[(size_t)b * H_ + c] = acc2;
}

// ---------------- per-graph FiLM params: gb = cond @ film_w + film_b (per layer) ---
__global__ void k_film(const float* __restrict__ cond, const float* __restrict__ fw,
                       const float* __restrict__ fb, float* __restrict__ gb) {
    __shared__ float cs[256];
    int bi = blockIdx.x;              // 0 .. D_*N_BATCH-1
    int l = bi >> 10;                 // N_BATCH == 1024
    int b = bi & 1023;
    int c = threadIdx.x;
    cs[c] = cond[(size_t)b * H_ + c];
    __syncthreads();
    const float* w = fw + (size_t)l * H_ * 2 * H_;
    float a0 = fb[(size_t)l * 2 * H_ + c];
    float a1 = fb[(size_t)l * 2 * H_ + H_ + c];
    for (int k = 0; k < 256; ++k) {
        float cv = cs[k];
        a0 += cv * w[(size_t)k * 2 * H_ + c];
        a1 += cv * w[(size_t)k * 2 * H_ + H_ + c];
    }
    size_t base = ((size_t)l * N_BATCH + b) * 2 * H_;
    gb[base + c] = a0;
    gb[base + H_ + c] = a1;
}

// ---------------- convert conv2_w (f32 [l][k][n]) -> bf16 transposed [l][n][k] ----
__global__ void k_wconv(const float* __restrict__ w, __bf16* __restrict__ wt, int total) {
    int g = blockIdx.x * 256 + threadIdx.x;
    if (g >= total) return;
    int l = g >> 16;
    int r = g & 65535;
    int n = r >> 8, k = r & 255;
    wt[((size_t)l << 16) + (n << 8) + k] = (__bf16)w[((size_t)l << 16) + (k << 8) + n];
}

// ---------------- input projection: h = (aggx*ni)@Wi + bi + e_mean ----------------
__global__ void k_input(const float* __restrict__ aggx, const float* __restrict__ es16,
                        const float* __restrict__ ni_v, const float* __restrict__ degi,
                        const float* __restrict__ wi, const float* __restrict__ bi,
                        const float* __restrict__ we, const float* __restrict__ be,
                        float* __restrict__ h) {
    int v = blockIdx.x, c = threadIdx.x;
    float d0 = 0.0f;
    #pragma unroll
    for (int f = 0; f < 32; ++f) d0 += aggx[(size_t)v * 32 + f] * wi[f * H_ + c];
    float acc = d0 * ni_v[v] + bi[c];
    float de = degi[v];
    if (de > 0.0f) {
        float d1 = 0.0f;
        #pragma unroll
        for (int f = 0; f < 16; ++f) d1 += es16[(size_t)v * 16 + f] * we[f * H_ + c];
        acc += d1 / de + be[c];   // bias contributes only when in-degree > 0
    }
    h[(size_t)v * H_ + c] = acc;
}

// ---------------- per-layer edge scatter: agg[dst] += h[src]*no[src] ----------------
__global__ void k_scath(const float* __restrict__ h, const float* __restrict__ no_v,
                        const int* __restrict__ src, const int* __restrict__ dst,
                        float* __restrict__ agg) {
    int e = blockIdx.x;
    int c = threadIdx.x;
    int s = src[e];
    atomicAdd(&agg[(size_t)dst[e] * H_ + c], h[(size_t)s * H_ + c] * no_v[s]);
}

// ---------------- WMMA GEMM: y = bf16(agg * ni) @ Wt + bias ----------------
// Block: 128 thr (4 waves). Stripe: 16 rows x 256 cols. Each wave: 4 col tiles.
__global__ __launch_bounds__(128) void k_gemm(
    const float* __restrict__ agg, const float* __restrict__ ni_v,
    const __bf16* __restrict__ wt, const float* __restrict__ bias,
    float* __restrict__ y) {
    __shared__ __bf16 As[16 * LDA];
    const int t = threadIdx.x;
    const int row0 = blockIdx.x * 16;
    // stage A (16x256) into LDS as bf16 with fused ni row scale
    #pragma unroll
    for (int j = 0; j < 32; ++j) {
        int idx = t + 128 * j;
        int r = idx >> 8, k = idx & 255;
        As[r * LDA + k] = (__bf16)(agg[(size_t)(row0 + r) * H_ + k] * ni_v[row0 + r]);
    }
    __syncthreads();
    const int wave = t >> 5;
    const int lane = t & 31;
    const int nlo = lane & 15;
    const int kh  = lane >> 4;     // khalf per ISA A/B layout
    v8f acc[4] = {};
    #pragma unroll
    for (int kb = 0; kb < 8; ++kb) {
        union { v16bf v; unsigned int u[8]; } a;
        #pragma unroll
        for (int q = 0; q < 8; ++q) {
            int k = kb * 32 + ((q >> 2) << 4) + ((q & 3) << 1) + (kh << 3);
            a.u[q] = *(const unsigned int*)(&As[nlo * LDA + k]);
        }
        #pragma unroll
        for (int c = 0; c < 4; ++c) {
            int n = (wave * 4 + c) * 16 + nlo;
            union { v16bf v; unsigned int u[8]; } bfr;
            #pragma unroll
            for (int q = 0; q < 8; ++q) {
                int k = kb * 32 + ((q >> 2) << 4) + ((q & 3) << 1) + (kh << 3);
                bfr.u[q] = *(const unsigned int*)(&wt[n * H_ + k]);
            }
            acc[c] = __builtin_amdgcn_wmma_f32_16x16x32_bf16(
                false, a.v, false, bfr.v, (short)0, acc[c], false, false);
        }
    }
    // D layout: lane holds (M = q + 8*kh, N = nlo) for VGPR q
    #pragma unroll
    for (int c = 0; c < 4; ++c) {
        int n = (wave * 4 + c) * 16 + nlo;
        float bn = bias[n];
        #pragma unroll
        for (int q = 0; q < 8; ++q) {
            int m = q + 8 * kh;
            y[(size_t)(row0 + m) * H_ + n] = acc[c][q] + bn;
        }
    }
}

// ---------------- epilogue: LayerNorm -> FiLM -> SiLU -> residual ----------------
__global__ void k_epilogue(const float* __restrict__ y, float* __restrict__ h,
                           const float* __restrict__ gbuf, const int* __restrict__ n_index,
                           const float* __restrict__ g, const float* __restrict__ bvec) {
    __shared__ float red[256];
    int v = blockIdx.x, c = threadIdx.x;
    float x = y[(size_t)v * H_ + c];
    red[c] = x;
    __syncthreads();
    for (int s = 128; s > 0; s >>= 1) { if (c < s) red[c] += red[c + s]; __syncthreads(); }
    float mu = red[0] * (1.0f / 256.0f);
    __syncthreads();
    float d = x - mu;
    red[c] = d * d;
    __syncthreads();
    for (int s = 128; s > 0; s >>= 1) { if (c < s) red[c] += red[c + s]; __syncthreads(); }
    float var = red[0] * (1.0f / 256.0f);
    float xn = d * rsqrtf(var + 1e-5f) * g[c] + bvec[c];
    int gi = n_index[v];
    float gam = gbuf[(size_t)gi * 2 * H_ + c];
    float bet = gbuf[(size_t)gi * 2 * H_ + H_ + c];
    float z = xn * (1.0f + gam) + bet;
    float sl = z / (1.0f + __expf(-z));    // SiLU
    h[(size_t)v * H_ + c] += sl;
}

// ---------------- graph pooling + head ----------------
__global__ void k_pool(const float* __restrict__ h, const int* __restrict__ n_index,
                       float* __restrict__ gsum, float* __restrict__ gcnt) {
    int v = blockIdx.x, c = threadIdx.x;
    int gidx = n_index[v];
    atomicAdd(&gsum[(size_t)gidx * H_ + c], h[(size_t)v * H_ + c]);
    if (c == 0) atomicAdd(&gcnt[gidx], 1.0f);
}

__global__ void k_head(const float* __restrict__ gsum, const float* __restrict__ gcnt,
                       const float* __restrict__ hw, const float* __restrict__ hb,
                       float* __restrict__ out) {
    __shared__ float red[256];
    int b = blockIdx.x, c = threadIdx.x;
    float m = gsum[(size_t)b * H_ + c] / fmaxf(gcnt[b], 1.0f);
    red[c] = m * hw[c];
    __syncthreads();
    for (int s = 128; s > 0; s >>= 1) { if (c < s) red[c] += red[c + s]; __syncthreads(); }
    if (c == 0) out[b] = red[0] + hb[0];
}

// ---------------------------------------------------------------------------
extern "C" void kernel_launch(void* const* d_in, const int* in_sizes, int n_in,
                              void* d_out, int out_size, void* d_ws, size_t ws_size,
                              hipStream_t stream) {
    const float* node_x  = (const float*)d_in[0];
    const float* edge_e  = (const float*)d_in[1];
    const float* t_in    = (const float*)d_in[2];
    const int*   n_index = (const int*)d_in[3];
    const int*   src     = (const int*)d_in[4];
    const int*   dst     = (const int*)d_in[5];
    const float* in_w    = (const float*)d_in[6];
    const float* in_b    = (const float*)d_in[7];
    const float* e_w     = (const float*)d_in[8];
    const float* e_b     = (const float*)d_in[9];
    const float* t_w1    = (const float*)d_in[10];
    const float* t_b1    = (const float*)d_in[11];
    const float* t_w2    = (const float*)d_in[12];
    const float* t_b2    = (const float*)d_in[13];
    // conv1 branch is dead in the reference (DCE'd) -> skip d_in[14],[15],[18],[19],[22],[23]
    const float* conv2_w = (const float*)d_in[16];
    const float* conv2_b = (const float*)d_in[17];
    const float* ln2_g   = (const float*)d_in[20];
    const float* ln2_b   = (const float*)d_in[21];
    const float* film2_w = (const float*)d_in[24];
    const float* film2_b = (const float*)d_in[25];
    const float* head_w  = (const float*)d_in[26];
    const float* head_b  = (const float*)d_in[27];
    float* out = (float*)d_out;

    float* ws = (float*)d_ws;
    size_t off = 0;
    float* deg_out = ws + off; off += N_NODES;
    float* deg_in  = ws + off; off += N_NODES;
    float* no_v    = ws + off; off += N_NODES;
    float* ni_v    = ws + off; off += N_NODES;
    float* aggx    = ws + off; off += (size_t)N_NODES * 32;
    float* es16    = ws + off; off += (size_t)N_NODES * 16;
    float* cond    = ws + off; off += (size_t)N_BATCH * H_;
    float* filmgb  = ws + off; off += (size_t)D_ * N_BATCH * 2 * H_;
    float* gsum    = ws + off; off += (size_t)N_BATCH * H_;
    float* gcnt    = ws + off; off += N_BATCH;
    float* agg     = ws + off; off += (size_t)N_NODES * H_;
    float* ybuf    = ws + off; off += (size_t)N_NODES * H_;
    float* hbuf    = ws + off; off += (size_t)N_NODES * H_;
    __bf16* wbf    = (__bf16*)(ws + off);   // D_*H_*H_ bf16 (transposed weights)

    // zero accumulation buffers (deg_out|deg_in contiguous; aggx|es16 contiguous; gsum|gcnt contiguous)
    hipMemsetAsync(deg_out, 0, sizeof(float) * (size_t)2 * N_NODES, stream);
    hipMemsetAsync(aggx, 0, sizeof(float) * (size_t)N_NODES * 48, stream);
    hipMemsetAsync(gsum, 0, sizeof(float) * ((size_t)N_BATCH * H_ + N_BATCH), stream);

    k_deg<<<(N_EDGES + 255) / 256, 256, 0, stream>>>(src, dst, deg_out, deg_in, N_EDGES);
    k_norms<<<(N_NODES + 255) / 256, 256, 0, stream>>>(deg_out, deg_in, no_v, ni_v, N_NODES);
    k_scat32<<<(N_EDGES * 32 + 255) / 256, 256, 0, stream>>>(node_x, no_v, src, dst, aggx, N_EDGES * 32);
    k_scat16<<<(N_EDGES * 16 + 255) / 256, 256, 0, stream>>>(edge_e, dst, es16, N_EDGES * 16);
    k_cond<<<N_BATCH, 256, 0, stream>>>(t_in, t_w1, t_b1, t_w2, t_b2, cond);
    k_film<<<D_ * N_BATCH, 256, 0, stream>>>(cond, film2_w, film2_b, filmgb);
    k_wconv<<<(D_ * H_ * H_ + 255) / 256, 256, 0, stream>>>(conv2_w, wbf, D_ * H_ * H_);
    k_input<<<N_NODES, 256, 0, stream>>>(aggx, es16, ni_v, deg_in, in_w, in_b, e_w, e_b, hbuf);

    for (int l = 0; l < D_; ++l) {
        hipMemsetAsync(agg, 0, sizeof(float) * (size_t)N_NODES * H_, stream);
        k_scath<<<N_EDGES, 256, 0, stream>>>(hbuf, no_v, src, dst, agg);
        k_gemm<<<N_NODES / 16, 128, 0, stream>>>(agg, ni_v, wbf + (size_t)l * H_ * H_,
                                                 conv2_b + l * H_, ybuf);
        k_epilogue<<<N_NODES, 256, 0, stream>>>(ybuf, hbuf,
                                                filmgb + (size_t)l * N_BATCH * 2 * H_,
                                                n_index, ln2_g + l * H_, ln2_b + l * H_);
    }

    k_pool<<<N_NODES, 256, 0, stream>>>(hbuf, n_index, gsum, gcnt);
    k_head<<<N_BATCH, 256, 0, stream>>>(gsum, gcnt, head_w, head_b, out);
}
